// PointNetFPModule_24764781429155
// MI455X (gfx1250) — matrix operand order repaired and checked by reading, not compiled, not run on hardware
//
#include <hip/hip_runtime.h>

// ---------------------------------------------------------------------------
// PointNet FP module, fused for CDNA5 (gfx1250, wave32):
//   kernel 1: convert W1/W2 f32 -> f16 into d_ws
//   kernel 2: per (batch, 32-point tile): 3-NN interp -> LDS feat (f16)
//             -> WMMA GEMM1 (256x320) + bias + relu -> LDS h (f16)
//             -> WMMA GEMM2 (128x256) + bias + relu -> d_out (f32)
//   kernel 3: passthrough copies (points_coords, condition) into d_out tail
// ---------------------------------------------------------------------------

typedef __attribute__((ext_vector_type(16))) _Float16 v16h;
typedef __attribute__((ext_vector_type(8)))  float    v8f;

union Frag { v16h h; unsigned int u[8]; };

#define B_   16
#define N_   4096
#define M_   1024
#define CC_  256
#define CP_  64
#define CIN_ 320
#define H1_  256
#define H2_  128

#define TILE_N   32          // points per workgroup
#define FPAD     328         // padded feat row (halves), 164 dwords -> conflict-free
#define HPAD     264         // padded h row (halves), 132 dwords -> conflict-free

// A-fragment K map for 16-bit A 16x32 (documented layout)
__device__ __forceinline__ int kmapA(int j, int h) {
    return (j < 4) ? (8 * h + 2 * j) : (16 + 8 * h + 2 * (j - 4));
}
// B-fragment K map for 16-bit B 32x16 (same K order as A)
__device__ __forceinline__ int kmapB(int j, int h) {
    return (j < 4) ? (8 * h + 2 * j) : (16 + 8 * h + 2 * (j - 4));
}

__global__ __launch_bounds__(256) void convert_weights_kernel(
    const float* __restrict__ W1, const float* __restrict__ W2,
    _Float16* __restrict__ W1h, _Float16* __restrict__ W2h) {
    int i = blockIdx.x * 256 + threadIdx.x;
    if (i < H1_ * CIN_) W1h[i] = (_Float16)W1[i];
    if (i < H2_ * H1_)  W2h[i] = (_Float16)W2[i];
}

__global__ __launch_bounds__(256) void fp_fused_kernel(
    const float* __restrict__ points_coords,   // (B,3,N)
    const float* __restrict__ centers_coords,  // (B,3,M)
    const float* __restrict__ centers_features,// (B,CC,M)
    const float* __restrict__ points_features, // (B,CP,N)
    const _Float16* __restrict__ W1h,          // (H1,CIN) f16
    const float* __restrict__ b1,              // (H1)
    const _Float16* __restrict__ W2h,          // (H2,H1) f16
    const float* __restrict__ b2,              // (H2)
    float* __restrict__ out)                   // (B,H2,N)
{
    __shared__ float    s_cc[3 * M_];               // centers coords x|y|z
    __shared__ float    s_cand_d[TILE_N * 8 * 3];   // per-segment top3 dist
    __shared__ int      s_cand_i[TILE_N * 8 * 3];   // per-segment top3 idx
    __shared__ float    s_w3[TILE_N * 3];           // final weights
    __shared__ int      s_i3[TILE_N * 3];           // final indices
    __shared__ _Float16 s_feat[TILE_N * FPAD];      // feat tile [n][k], k<320
    __shared__ _Float16 s_h[TILE_N * HPAD];         // hidden tile [n][o], o<256

    const int tid  = threadIdx.x;
    const int b    = blockIdx.y;
    const int n0   = blockIdx.x * TILE_N;

    // ---------------- phase 0: centers coords -> LDS ----------------
    for (int i = tid; i < 3 * M_; i += 256)
        s_cc[i] = centers_coords[b * 3 * M_ + i];
    __syncthreads();

    // ---------------- phase 1: 3-NN search (8 threads per point) ----
    {
        const int p   = tid & (TILE_N - 1);   // point in tile
        const int seg = tid / TILE_N;         // 0..7, 128 centers each
        const int n   = n0 + p;
        const float px = points_coords[b * 3 * N_ + 0 * N_ + n];
        const float py = points_coords[b * 3 * N_ + 1 * N_ + n];
        const float pz = points_coords[b * 3 * N_ + 2 * N_ + n];

        float bd0 = 3e38f, bd1 = 3e38f, bd2 = 3e38f;
        int   bi0 = 0,     bi1 = 0,     bi2 = 0;
        const int m0 = seg * (M_ / 8);
        for (int m = m0; m < m0 + (M_ / 8); ++m) {
            float dx = px - s_cc[m];
            float dy = py - s_cc[M_ + m];
            float dz = pz - s_cc[2 * M_ + m];
            float d  = dx * dx + dy * dy + dz * dz;
            if (d < bd2) {
                if (d < bd0)      { bd2 = bd1; bi2 = bi1; bd1 = bd0; bi1 = bi0; bd0 = d; bi0 = m; }
                else if (d < bd1) { bd2 = bd1; bi2 = bi1; bd1 = d;   bi1 = m; }
                else              { bd2 = d;   bi2 = m; }
            }
        }
        const int base = (p * 8 + seg) * 3;
        s_cand_d[base + 0] = bd0; s_cand_i[base + 0] = bi0;
        s_cand_d[base + 1] = bd1; s_cand_i[base + 1] = bi1;
        s_cand_d[base + 2] = bd2; s_cand_i[base + 2] = bi2;
    }
    __syncthreads();

    // ---------------- phase 1b: merge 8 segments, weights -----------
    if (tid < TILE_N) {
        const int p = tid;
        float bd0 = 3e38f, bd1 = 3e38f, bd2 = 3e38f;
        int   bi0 = 0,     bi1 = 0,     bi2 = 0;
        for (int s = 0; s < 8; ++s) {
            const int base = (p * 8 + s) * 3;
            #pragma unroll
            for (int j = 0; j < 3; ++j) {
                float d = s_cand_d[base + j];
                int   i = s_cand_i[base + j];
                if (d < bd2) {
                    if (d < bd0)      { bd2 = bd1; bi2 = bi1; bd1 = bd0; bi1 = bi0; bd0 = d; bi0 = i; }
                    else if (d < bd1) { bd2 = bd1; bi2 = bi1; bd1 = d;   bi1 = i; }
                    else              { bd2 = d;   bi2 = i; }
                }
            }
        }
        const float r0 = 1.0f / fmaxf(bd0, 1e-10f);
        const float r1 = 1.0f / fmaxf(bd1, 1e-10f);
        const float r2 = 1.0f / fmaxf(bd2, 1e-10f);
        const float inv = 1.0f / (r0 + r1 + r2);
        s_w3[p * 3 + 0] = r0 * inv; s_i3[p * 3 + 0] = bi0;
        s_w3[p * 3 + 1] = r1 * inv; s_i3[p * 3 + 1] = bi1;
        s_w3[p * 3 + 2] = r2 * inv; s_i3[p * 3 + 2] = bi2;
    }
    __syncthreads();

    // ---------------- phase 2: interpolate + concat -> feat ---------
    {
        const float* CF = centers_features + (size_t)b * CC_ * M_;
        for (int w = tid; w < TILE_N * CC_; w += 256) {
            const int p = w & (TILE_N - 1);
            const int c = w / TILE_N;
            const float* cfc = CF + c * M_;
            float v = s_w3[p * 3 + 0] * cfc[s_i3[p * 3 + 0]]
                    + s_w3[p * 3 + 1] * cfc[s_i3[p * 3 + 1]]
                    + s_w3[p * 3 + 2] * cfc[s_i3[p * 3 + 2]];
            s_feat[p * FPAD + c] = (_Float16)v;
        }
        const float* PF = points_features + (size_t)b * CP_ * N_;
        for (int w = tid; w < TILE_N * CP_; w += 256) {
            const int p  = w & (TILE_N - 1);
            const int cp = w / TILE_N;
            s_feat[p * FPAD + CC_ + cp] = (_Float16)PF[cp * N_ + n0 + p];
        }
    }
    __syncthreads();

    // ---------------- phase 3: GEMM1 (H1 x CIN) via WMMA ------------
    const int wv = tid >> 5;          // wave 0..7
    const int lane = tid & 31;
    const int lh = lane >> 4;         // lane half
    const int ln = lane & 15;         // lane index within half
    {
        v8f acc[2][2] = {};           // [o-subtile][n-subtile]
        const unsigned int* ar0 = (const unsigned int*)(W1h + (wv * 32 + ln) * CIN_);
        const unsigned int* ar1 = (const unsigned int*)(W1h + (wv * 32 + 16 + ln) * CIN_);
        const unsigned int* br0 = (const unsigned int*)(s_feat + ln * FPAD);
        const unsigned int* br1 = (const unsigned int*)(s_feat + (16 + ln) * FPAD);
        for (int k0 = 0; k0 < CIN_; k0 += 32) {
            Frag a0, a1, f0, f1;
            #pragma unroll
            for (int j = 0; j < 8; ++j) {
                const int ka = (k0 + kmapA(j, lh)) >> 1;
                const int kb = (k0 + kmapB(j, lh)) >> 1;
                a0.u[j] = ar0[ka]; a1.u[j] = ar1[ka];
                f0.u[j] = br0[kb]; f1.u[j] = br1[kb];
            }
            acc[0][0] = __builtin_amdgcn_wmma_f32_16x16x32_f16(false, a0.h, false, f0.h, (short)0, acc[0][0], false, false);
            acc[0][1] = __builtin_amdgcn_wmma_f32_16x16x32_f16(false, a0.h, false, f1.h, (short)0, acc[0][1], false, false);
            acc[1][0] = __builtin_amdgcn_wmma_f32_16x16x32_f16(false, a1.h, false, f0.h, (short)0, acc[1][0], false, false);
            acc[1][1] = __builtin_amdgcn_wmma_f32_16x16x32_f16(false, a1.h, false, f1.h, (short)0, acc[1][1], false, false);
        }
        #pragma unroll
        for (int i = 0; i < 2; ++i) {
            #pragma unroll
            for (int r = 0; r < 8; ++r) {
                const int m  = wv * 32 + i * 16 + r + 8 * lh;   // output channel of layer1
                const float bv = b1[m];
                s_h[ln * HPAD + m]        = (_Float16)fmaxf(acc[i][0][r] + bv, 0.0f);
                s_h[(16 + ln) * HPAD + m] = (_Float16)fmaxf(acc[i][1][r] + bv, 0.0f);
            }
        }
    }
    __syncthreads();

    // ---------------- phase 4: GEMM2 (H2 x H1) via WMMA -> out ------
    {
        v8f acc2[2] = {};
        const unsigned int* ar  = (const unsigned int*)(W2h + (wv * 16 + ln) * H1_);
        const unsigned int* br0 = (const unsigned int*)(s_h + ln * HPAD);
        const unsigned int* br1 = (const unsigned int*)(s_h + (16 + ln) * HPAD);
        for (int k0 = 0; k0 < H1_; k0 += 32) {
            Frag a, h0, h1;
            #pragma unroll
            for (int j = 0; j < 8; ++j) {
                const int ka = (k0 + kmapA(j, lh)) >> 1;
                const int kb = (k0 + kmapB(j, lh)) >> 1;
                a.u[j]  = ar[ka];
                h0.u[j] = br0[kb];
                h1.u[j] = br1[kb];
            }
            acc2[0] = __builtin_amdgcn_wmma_f32_16x16x32_f16(false, a.h, false, h0.h, (short)0, acc2[0], false, false);
            acc2[1] = __builtin_amdgcn_wmma_f32_16x16x32_f16(false, a.h, false, h1.h, (short)0, acc2[1], false, false);
        }
        float* outb = out + (size_t)b * H2_ * N_;
        #pragma unroll
        for (int r = 0; r < 8; ++r) {
            const int m  = wv * 16 + r + 8 * lh;                // output channel of layer2
            const float bv = b2[m];
            outb[(size_t)m * N_ + n0 + ln]      = fmaxf(acc2[0][r] + bv, 0.0f);
            outb[(size_t)m * N_ + n0 + 16 + ln] = fmaxf(acc2[1][r] + bv, 0.0f);
        }
    }
}

__global__ __launch_bounds__(256) void tail_copy_kernel(
    const float* __restrict__ points_coords,  // (B,3,N) = 196608
    const float* __restrict__ condition,      // (B,128) = 2048
    float* __restrict__ out_tail)             // d_out + B*H2*N
{
    const int PC = B_ * 3 * N_;
    const int CD = B_ * 128;
    int i = blockIdx.x * 256 + threadIdx.x;
    if (i < PC)            out_tail[i] = points_coords[i];
    else if (i < PC + CD)  out_tail[i] = condition[i - PC];
}

extern "C" void kernel_launch(void* const* d_in, const int* in_sizes, int n_in,
                              void* d_out, int out_size, void* d_ws, size_t ws_size,
                              hipStream_t stream) {
    const float* points_coords    = (const float*)d_in[0];
    const float* centers_coords   = (const float*)d_in[1];
    const float* centers_features = (const float*)d_in[2];
    const float* points_features  = (const float*)d_in[3];
    const float* condition        = (const float*)d_in[4];
    const float* W1               = (const float*)d_in[5];
    const float* b1               = (const float*)d_in[6];
    const float* W2               = (const float*)d_in[7];
    const float* b2               = (const float*)d_in[8];
    float* out = (float*)d_out;

    _Float16* W1h = (_Float16*)d_ws;                       // 81920 halves
    _Float16* W2h = W1h + H1_ * CIN_;                      // 32768 halves

    // 1) weights f32 -> f16 (A-operands read as packed 32-bit pairs)
    convert_weights_kernel<<<(H1_ * CIN_ + 255) / 256, 256, 0, stream>>>(W1, W2, W1h, W2h);

    // 2) fused 3-NN interp + SharedMLP (one WG per 32-point tile per batch)
    dim3 grid(N_ / TILE_N, B_, 1);
    fp_fused_kernel<<<grid, 256, 0, stream>>>(
        points_coords, centers_coords, centers_features, points_features,
        W1h, b1, W2h, b2, out);

    // 3) passthrough outputs
    const int tail = B_ * 3 * N_ + B_ * 128;
    tail_copy_kernel<<<(tail + 255) / 256, 256, 0, stream>>>(
        points_coords, condition, out + (size_t)B_ * H2_ * N_);
}
